// NeRFAccField_9680856285274
// MI455X (gfx1250) — matrix operand order repaired and checked by reading, not compiled, not run on hardware
//
#include <hip/hip_runtime.h>
#include <hip/hip_fp16.h>
#include <hip/hip_bf16.h>

typedef __attribute__((ext_vector_type(16))) _Float16 v16h;
typedef __attribute__((ext_vector_type(8)))  float    v8f;

#define NL 16
#define T_SIZE (1u << 19)

// scale_l = 16 * 1.447269237440378^l - 1 ; res_l = ceil(scale_l)+1
__constant__ float c_scale[NL] = {
    15.0f,         22.1563078f,  32.5134162f,  47.5029426f,
    69.1968171f,   100.5936880f, 146.0334224f, 211.7969488f,
    306.9744784f,  444.7219856f, 644.0797152f, 932.6040288f,
    1350.1763952f, 1954.5160080f, 2829.1581584f, 4095.0008304f};
__constant__ int c_res[NL] = {16,  24,  34,  49,  71,   102,  148,  213,
                              308, 446, 646, 934, 1352, 1956, 2831, 4097};

union V16H {
    v16h v;
    _Float16 h[16];
    uint4 q[2];
};
union H2U {
    _Float16 h[2];
    unsigned u;
};

__global__ __launch_bounds__(256) void nerf_fused(
    const float* __restrict__ points,   // [N,3]
    const float* __restrict__ table,    // [16, T, 2]
    const float* __restrict__ dw1,      // [32,64]
    const float* __restrict__ dw2,      // [64,1]
    const float* __restrict__ fw1,      // [32,64]
    const float* __restrict__ fw2,      // [64,3]
    float* __restrict__ out_color,      // [N,3]
    float* __restrict__ out_density,    // [N]
    int N) {
    // Feats in A-friendly row layout; weights pre-swizzled into WMMA B layout:
    //   B lane l: col n = l&15, K groups kb..kb+7, kb+16..kb+23 (kb = 8*(l>>4)), as A
    __shared__ __align__(16) _Float16 sFeat[256][32];      // 16 KB
    __shared__ __align__(16) _Float16 sW1B[2][4][16][32];  // 8 KB [net][t][n][k]
    __shared__ __align__(16) _Float16 sW2B[2][16][64];     // 4 KB [net][n][k]
    __shared__ __align__(16) _Float16 sHid[8][16][64];     // 16 KB per-wave

    const int tid  = threadIdx.x;
    const int lane = tid & 31;
    const int wave = tid >> 5;
    const int blockBase = blockIdx.x * 256;
    const int pt = blockBase + tid;

    // ---- cooperative weight load + convert + B-layout swizzle ----
    for (int i = tid; i < 32 * 64; i += 256) {
        const int k = i >> 6;          // enc-dim K (0..31)
        const int c = i & 63;          // hidden col (0..63)
        const int t = c >> 4, n = c & 15;
        sW1B[0][t][n][k] = (_Float16)dw1[i];
        sW1B[1][t][n][k] = (_Float16)fw1[i];
    }
    for (int i = tid; i < 16 * 64; i += 256) {
        const int n = i >> 6;          // out col (0..15)
        const int k = i & 63;          // hidden K (0..63)
        sW2B[0][n][k] = (n == 0) ? (_Float16)dw2[k] : (_Float16)0.0f;
        sW2B[1][n][k] = (n < 3) ? (_Float16)fw2[k * 3 + n] : (_Float16)0.0f;
    }

    // ---- hash-grid encode: one point per lane, packed f16 pairs -> LDS ----
    {
        const int ptc = (pt < N) ? pt : (N - 1);
        const float x0 = points[3 * ptc + 0];
        const float y0 = points[3 * ptc + 1];
        const float z0 = points[3 * ptc + 2];
        const float x = fminf(fmaxf((x0 + 1.0f) * 0.5f, 0.0f), 1.0f);
        const float y = fminf(fmaxf((y0 + 1.0f) * 0.5f, 0.0f), 1.0f);
        const float z = fminf(fmaxf((z0 + 1.0f) * 0.5f, 0.0f), 1.0f);

        const float2* __restrict__ tbl2 = (const float2*)table;
        unsigned* featRow = (unsigned*)&sFeat[tid][0];

        for (int l = 0; l < NL; ++l) {
            const float scale = c_scale[l];
            const int res = c_res[l];
            const float px = x * scale + 0.5f;
            const float py = y * scale + 0.5f;
            const float pz = z * scale + 0.5f;
            const float fx = floorf(px), fy = floorf(py), fz = floorf(pz);
            const float wx = px - fx, wy = py - fy, wz = pz - fz;
            const int ix = (int)fx, iy = (int)fy, iz = (int)fz;
            const bool dense = (l < 5);  // res^3 <= 2^19 for levels 0..4

            float f0 = 0.0f, f1 = 0.0f;
            const size_t lvlBase = (size_t)l * (size_t)T_SIZE;
#pragma unroll
            for (int c = 0; c < 8; ++c) {
                const int ox = (c >> 2) & 1, oy = (c >> 1) & 1, oz = c & 1;
                const int cx = min(max(ix + ox, 0), res - 1);
                const int cy = min(max(iy + oy, 0), res - 1);
                const int cz = min(max(iz + oz, 0), res - 1);
                unsigned idx;
                if (dense) {
                    idx = (unsigned)(cx + res * (cy + res * cz));
                } else {
                    idx = ((unsigned)cx * 1u ^ (unsigned)cy * 2654435761u ^
                           (unsigned)cz * 805459861u) & (T_SIZE - 1u);
                }
                const float w = (ox ? wx : 1.0f - wx) * (oy ? wy : 1.0f - wy) *
                                (oz ? wz : 1.0f - wz);
                const float2 g = tbl2[lvlBase + idx];
                f0 = fmaf(w, g.x, f0);
                f1 = fmaf(w, g.y, f1);
            }
            H2U p2;
            p2.h[0] = (_Float16)f0;
            p2.h[1] = (_Float16)f1;
            featRow[l] = p2.u;  // one ds_store_b32 per level
        }
    }

    __syncthreads();

    // ---- fused MLPs via WMMA; each wave owns 32 points (2 M-tiles) ----
    const int n  = lane & 15;   // output column / A row
    const int hi = lane >> 4;   // half-wave select
    const int kb = hi ? 8 : 0;  // A layout K groups {kb..kb+7, kb+16..kb+23}
    const int kB = hi ? 16 : 0; // unused; B loads use the kb groups too

    for (int net = 0; net < 2; ++net) {
        // B operands: loaded once per net, held in VGPRs across both M-tiles
        V16H B1[4], B2[2];
#pragma unroll
        for (int t = 0; t < 4; ++t) {
            const _Float16* bsrc = &sW1B[net][t][n][kb];
            B1[t].q[0] = *(const uint4*)(bsrc);
            B1[t].q[1] = *(const uint4*)(bsrc + 16);
        }
#pragma unroll
        for (int kc = 0; kc < 2; ++kc) {
            const _Float16* bsrc = &sW2B[net][n][kc * 32 + kb];
            B2[kc].q[0] = *(const uint4*)(bsrc);
            B2[kc].q[1] = *(const uint4*)(bsrc + 16);
        }

        for (int tile = 0; tile < 2; ++tile) {
            const int rowBase = wave * 32 + tile * 16;

            V16H A;
            {
                const _Float16* arow = &sFeat[rowBase + n][0];
                A.q[0] = *(const uint4*)(arow + kb);
                A.q[1] = *(const uint4*)(arow + kb + 16);
            }

            // layer 1: hid[16x64] = relu(A[16x32] x W1[32x64])
#pragma unroll
            for (int t = 0; t < 4; ++t) {
                v8f acc = {};
                acc = __builtin_amdgcn_wmma_f32_16x16x32_f16(
                    false, A.v, false, B1[t].v, (short)0, acc, false, false);
#pragma unroll
                for (int j = 0; j < 8; ++j) {
                    float v = acc[j] > 0.0f ? acc[j] : 0.0f;
                    sHid[wave][j + hi * 8][t * 16 + n] = (_Float16)v;
                }
            }
            // per-wave scratch: same-wave DS ordering covers the hazard

            // layer 2: out[16x16] = hid[16x64] x W2[64x16], K split 2x32
            v8f acc2 = {};
#pragma unroll
            for (int kc = 0; kc < 2; ++kc) {
                V16H A2;
                const _Float16* hrow = &sHid[wave][n][0];
                A2.q[0] = *(const uint4*)(hrow + kc * 32 + kb);
                A2.q[1] = *(const uint4*)(hrow + kc * 32 + kb + 16);
                acc2 = __builtin_amdgcn_wmma_f32_16x16x32_f16(
                    false, A2.v, false, B2[kc].v, (short)0, acc2, false, false);
            }

            // activations + scattered stores (D layout direct)
            if (n < 3) {
#pragma unroll
                for (int j = 0; j < 8; ++j) {
                    const int m = j + hi * 8;
                    const int pidx = blockBase + rowBase + m;
                    if (pidx < N) {
                        const float v = acc2[j];
                        if (net == 0) {
                            if (n == 0) {  // density = softplus
                                out_density[pidx] =
                                    (v > 20.0f) ? v : log1pf(expf(v));
                            }
                        } else {           // color = sigmoid
                            out_color[pidx * 3 + n] = 1.0f / (1.0f + expf(-v));
                        }
                    }
                }
            }
        }
    }
}

extern "C" void kernel_launch(void* const* d_in, const int* in_sizes, int n_in,
                              void* d_out, int out_size, void* d_ws,
                              size_t ws_size, hipStream_t stream) {
    const float* points = (const float*)d_in[0];
    const float* table  = (const float*)d_in[1];
    const float* dw1    = (const float*)d_in[2];
    const float* dw2    = (const float*)d_in[3];
    const float* fw1    = (const float*)d_in[4];
    const float* fw2    = (const float*)d_in[5];
    const int N = in_sizes[0] / 3;

    float* out_color   = (float*)d_out;              // [N,3] first
    float* out_density = out_color + (size_t)3 * N;  // [N] second

    const int blocks = (N + 255) / 256;
    nerf_fused<<<blocks, 256, 0, stream>>>(points, table, dw1, dw2, fw1, fw2,
                                           out_color, out_density, N);
}
